// GRSA_15977278341914
// MI455X (gfx1250) — compile-verified, matrix operand order checked by reading, not run on hardware
//
#include <hip/hip_runtime.h>
#include <hip/hip_bf16.h>

typedef _Float16 f16;
typedef __attribute__((ext_vector_type(16))) _Float16 v16h;
typedef __attribute__((ext_vector_type(8)))  _Float16 v8h;
typedef __attribute__((ext_vector_type(8)))  float    v8f;

#define LOG2E 1.44269504088896340736f

// ---------------------------------------------------------------------------
// WMMA helpers (CDNA5 wave32, V_WMMA_F32_16X16X32_F16)
// A/B register tile (16x32 f16, row-major source, row stride `stride` halves):
//   lane L<16  holds row (L)    , K = {koff..koff+7, koff+16..koff+23}, koff=0
//   lane L>=16 holds row (L-16) , koff=8
// which matches the ISA 16-bit A layout; B uses the identical register image
// with "row" meaning the B column (we feed k rows / v^T rows directly).
// C/D tile: VGPR r, lanes 0-15 -> (M=r, N=lane); lanes 16-31 -> (M=r+8, N=lane-16)
// ---------------------------------------------------------------------------
static __device__ __forceinline__ v16h load_tile_h(const f16* base, int stride, int lane) {
  const f16* p = base + (lane & 15) * stride + ((lane >> 4) << 3);
  v8h lo = *(const v8h*)(p);        // K koff..koff+7      (16B aligned)
  v8h hi = *(const v8h*)(p + 16);   // K koff+16..koff+23  (16B aligned)
  v16h r;
#pragma unroll
  for (int i = 0; i < 8; ++i) { r[i] = lo[i]; r[i + 8] = hi[i]; }
  return r;
}

static __device__ __forceinline__ v16h load_tile_f32cvt(const float* base, int stride, int lane) {
  const float* p = base + (lane & 15) * stride + ((lane >> 4) << 3);
  v16h r;
#pragma unroll
  for (int i = 0; i < 8; ++i) { r[i] = (f16)p[i]; r[i + 8] = (f16)p[16 + i]; }
  return r;
}

static __device__ __forceinline__ v8f wmma_f16(v16h a, v16h b, v8f c) {
  return __builtin_amdgcn_wmma_f32_16x16x32_f16(false, a, false, b, (short)0, c, false, false);
}

// Reductions across the 16 lanes of a half-wave (xor masks < 16 stay in-half).
static __device__ __forceinline__ float hred_max(float x) {
#pragma unroll
  for (int m = 1; m < 16; m <<= 1) x = fmaxf(x, __shfl_xor(x, m, 32));
  return x;
}
static __device__ __forceinline__ float hred_sum(float x) {
#pragma unroll
  for (int m = 1; m < 16; m <<= 1) x += __shfl_xor(x, m, 32);
  return x;
}

// ---------------------------------------------------------------------------
// Stage 0: NCHW f32 -> (b, N, c) f16
// ---------------------------------------------------------------------------
__global__ void __launch_bounds__(256) transpose_kernel(const float* __restrict__ x,
                                                        f16* __restrict__ x3h) {
  int i = blockIdx.x * 256 + threadIdx.x;      // 16*256*1024 elements
  if (i >= 16 * 256 * 1024) return;
  int p  = i & 1023;
  int ch = (i >> 10) & 255;
  int b  = i >> 18;
  x3h[((size_t)(b * 1024 + p)) * 256 + ch] = (f16)x[i];
}

// ---------------------------------------------------------------------------
// Stage 1: relative-position bias table: tab_sig[head*3969 + i] = 16*sigmoid(mlp(coords_i))
// ---------------------------------------------------------------------------
__global__ void __launch_bounds__(256) bias_table_kernel(const float* __restrict__ w1,
                                                         const float* __restrict__ b1,
                                                         const float* __restrict__ w2,
                                                         float* __restrict__ tab_sig) {
  int t = blockIdx.x * 256 + threadIdx.x;      // 8 * 3969
  if (t >= 8 * 3969) return;
  int i = t % 3969, head = t / 3969;
  int gy = i / 63, gx = i % 63;
  float ty = ((float)(gy - 31) / (31.0f + 1e-8f)) * 3.2f;
  float tx = ((float)(gx - 31) / (31.0f + 1e-8f)) * 3.2f;
  float sy = (ty > 0.f) ? 1.f : ((ty < 0.f) ? -1.f : 0.f);
  float sx = (tx > 0.f) ? 1.f : ((tx < 0.f) ? -1.f : 0.f);
  ty = sy * (1.0f - __expf(-fabsf(ty)));
  tx = sx * (1.0f - __expf(-fabsf(tx)));
  float acc = 0.f;
#pragma unroll 8
  for (int j = 0; j < 128; ++j) {
    float hj = ty * w1[j * 2 + 0] + tx * w1[j * 2 + 1] + b1[j];
    hj = fmaxf(hj, 0.f);
    acc += hj * w2[head * 128 + j];
  }
  tab_sig[t] = 16.0f / (1.0f + __expf(-acc));
}

// ---------------------------------------------------------------------------
// Stage 2/4: half-GEMM (rows = 16384 tokens, 128x128 weight for one half).
//  MODE 0: q/k  -> residual + per-head L2 norm, f16 out  ((b*8+head)*1024+n)*32+d
//  MODE 1: v    -> residual, f16 out TRANSPOSED          ((b*8+head)*32+d)*1024+n
//  MODE 2: proj -> +bias, f32 NCHW out                   (b*256+ch)*1024+n
// 8 waves, each owns a 16-row strip of the 128-row block; K loop = 4 x 32.
// ---------------------------------------------------------------------------
template <int MODE>
__global__ void __launch_bounds__(256) half_gemm_kernel(const f16* __restrict__ Xh,
                                                        const float* __restrict__ Xf,
                                                        const float* __restrict__ W,
                                                        const float* __restrict__ bvec,
                                                        f16* __restrict__ out_h,
                                                        float* __restrict__ out_f,
                                                        int half) {
  int tid = threadIdx.x, wave = tid >> 5, lane = tid & 31;
  int row0  = blockIdx.x * 128 + wave * 16;
  int nloc  = lane & 15;
  int mbase = (lane >> 4) << 3;

  v16h a[4];
#pragma unroll
  for (int kt = 0; kt < 4; ++kt) {
    if (MODE == 2) a[kt] = load_tile_f32cvt(Xf + (size_t)row0 * 256 + half * 128 + kt * 32, 256, lane);
    else           a[kt] = load_tile_h    (Xh + (size_t)row0 * 256 + half * 128 + kt * 32, 256, lane);
  }

#pragma unroll
  for (int j = 0; j < 4; ++j) {            // output cols 32j .. 32j+31 (= head j of this half)
    v8f acc0 = {}, acc1 = {};
#pragma unroll
    for (int kt = 0; kt < 4; ++kt) {
      // B column n = W[col, :] row (y = x @ W^T), f32 -> f16 on load
      v16h b0 = load_tile_f32cvt(W + (32 * j + 0)  * 128 + kt * 32, 128, lane);
      v16h b1 = load_tile_f32cvt(W + (32 * j + 16) * 128 + kt * 32, 128, lane);
      acc0 = wmma_f16(a[kt], b0, acc0);
      acc1 = wmma_f16(a[kt], b1, acc1);
    }
    int c0 = 32 * j + nloc, c1 = c0 + 16;
    float r0[8], r1[8];
#pragma unroll
    for (int r = 0; r < 8; ++r) {
      int row = row0 + mbase + r;
      float y0 = acc0[r] + bvec[c0];
      float y1 = acc1[r] + bvec[c1];
      if (MODE != 2) {                     // residual x + lin(x)
        y0 += (float)Xh[(size_t)row * 256 + half * 128 + c0];
        y1 += (float)Xh[(size_t)row * 256 + half * 128 + c1];
      }
      r0[r] = y0; r1[r] = y1;
    }
    if (MODE == 0) {                       // per-head (32-dim) L2 normalization
#pragma unroll
      for (int r = 0; r < 8; ++r) {
        float ss  = hred_sum(r0[r] * r0[r] + r1[r] * r1[r]);
        float inv = 1.0f / fmaxf(__fsqrt_rn(ss), 1e-12f);
        r0[r] *= inv; r1[r] *= inv;
      }
    }
#pragma unroll
    for (int r = 0; r < 8; ++r) {
      int row = row0 + mbase + r;
      int b = row >> 10, n = row & 1023;
      int head = half * 4 + j;
      if (MODE == 2) {
        out_f[((size_t)(b * 256 + half * 128 + c0)) * 1024 + n] = r0[r];
        out_f[((size_t)(b * 256 + half * 128 + c1)) * 1024 + n] = r1[r];
      } else if (MODE == 1) {
        out_h[((size_t)((b * 8 + head) * 32 + nloc))      * 1024 + n] = (f16)r0[r];
        out_h[((size_t)((b * 8 + head) * 32 + nloc + 16)) * 1024 + n] = (f16)r1[r];
      } else {
        out_h[((size_t)((b * 8 + head) * 1024 + n)) * 32 + nloc]      = (f16)r0[r];
        out_h[((size_t)((b * 8 + head) * 1024 + n)) * 32 + nloc + 16] = (f16)r1[r];
      }
    }
  }
}

// ---------------------------------------------------------------------------
// Stage 3: flash attention per (b, head, 128-row block). 8 waves x 16 q-rows.
// Never materializes the 1024x1024 attention matrix (512MB -> HBM-bound trap).
// ---------------------------------------------------------------------------
__global__ void __launch_bounds__(256) attn_kernel(const f16* __restrict__ q_ws,
                                                   const f16* __restrict__ k_ws,
                                                   const f16* __restrict__ vt_ws,
                                                   const float* __restrict__ tab_sig,
                                                   const float* __restrict__ logit_scale,
                                                   float* __restrict__ o_ws) {
  __shared__ float s_bias[3969];           // 16*sigmoid table slice for this head
  __shared__ f16   s_p[8 * 16 * 32];       // per-wave P staging (D-layout -> A-layout)

  int tid = threadIdx.x, wave = tid >> 5, lane = tid & 31;
  int rb = blockIdx.x, h = blockIdx.y, b = blockIdx.z;

  for (int i = tid; i < 3969; i += 256) s_bias[i] = tab_sig[h * 3969 + i];
  __syncthreads();

  float scale = __expf(fminf(logit_scale[h], 4.6051701859880914f)); // log(100)
  int row0  = rb * 128 + wave * 16;
  int nloc  = lane & 15;
  int mbase = (lane >> 4) << 3;

  const f16* qbh  = q_ws  + ((size_t)(b * 8 + h) * 1024) * 32;
  const f16* kbh  = k_ws  + ((size_t)(b * 8 + h) * 1024) * 32;
  const f16* vtbh = vt_ws + ((size_t)(b * 8 + h) * 32) * 1024;

  v16h qa = load_tile_h(qbh + row0 * 32, 32, lane);   // resident for whole loop

  int ry[8], rx[8];
#pragma unroll
  for (int r = 0; r < 8; ++r) { int row = row0 + mbase + r; ry[r] = row >> 5; rx[r] = row & 31; }

  v8f o0 = {}, o1 = {};
  float m_r[8], l_r[8];
#pragma unroll
  for (int r = 0; r < 8; ++r) { m_r[r] = -1e30f; l_r[r] = 0.f; }

  f16* pp = &s_p[wave * 512];

  for (int kb = 0; kb < 1024; kb += 32) {
    // ---- prefetch next iteration's K / V^T tiles (global_prefetch_b8) ----
    if (kb + 32 < 1024) {
      __builtin_prefetch(kbh + (size_t)(kb + 32) * 32 + (lane & 15) * 32, 0, 3);
      __builtin_prefetch(vtbh + (kb + 32) + (lane & 15) * 1024, 0, 3);
      __builtin_prefetch(vtbh + 16 * 1024 + (kb + 32) + (lane & 15) * 1024, 0, 3);
    }

    // ---- S = qn @ kn^T (two 16x16 tiles, K = hd = 32 in one WMMA each) ----
    v16h kt0 = load_tile_h(kbh + (size_t)kb * 32, 32, lane);
    v16h kt1 = load_tile_h(kbh + (size_t)(kb + 16) * 32, 32, lane);
    v8f z = {};
    v8f s0 = wmma_f16(qa, kt0, z);
    v8f s1 = wmma_f16(qa, kt1, z);

    int c0 = kb + nloc, c1 = c0 + 16;
    int cy0 = c0 >> 5, cx0 = c0 & 31, cy1 = c1 >> 5, cx1 = c1 & 31;

    float p0[8], p1[8];
#pragma unroll
    for (int r = 0; r < 8; ++r) {          // scale + gathered relative-position bias
      p0[r] = s0[r] * scale + s_bias[(ry[r] - cy0 + 31) * 63 + (rx[r] - cx0 + 31)];
      p1[r] = s1[r] * scale + s_bias[(ry[r] - cy1 + 31) * 63 + (rx[r] - cx1 + 31)];
    }

    // ---- online softmax (row stats per VGPR, reduced across the 16-lane half) ----
#pragma unroll
    for (int r = 0; r < 8; ++r) {
      float tmax  = hred_max(fmaxf(p0[r], p1[r]));
      float mnew  = fmaxf(m_r[r], tmax);
      float alpha = exp2f((m_r[r] - mnew) * LOG2E);
      m_r[r] = mnew;
      float e0 = exp2f((p0[r] - mnew) * LOG2E);
      float e1 = exp2f((p1[r] - mnew) * LOG2E);
      p0[r] = e0; p1[r] = e1;
      l_r[r] = l_r[r] * alpha + hred_sum(e0 + e1);
      o0[r] *= alpha; o1[r] *= alpha;
    }

    // ---- D-layout -> A-layout via per-wave LDS (same-wave DS ops are in-order) ----
#pragma unroll
    for (int r = 0; r < 8; ++r) {
      pp[(mbase + r) * 32 + nloc]      = (f16)p0[r];
      pp[(mbase + r) * 32 + nloc + 16] = (f16)p1[r];
    }
    v16h pa = load_tile_h(pp, 32, lane);

    // ---- O += P @ V (v stored transposed: vt[dim][key] so B loads are contiguous) ----
    v16h vb0 = load_tile_h(vtbh + kb, 1024, lane);             // dims 0..15
    v16h vb1 = load_tile_h(vtbh + 16 * 1024 + kb, 1024, lane); // dims 16..31
    o0 = wmma_f16(pa, vb0, o0);
    o1 = wmma_f16(pa, vb1, o1);
  }

  // ---- epilogue: normalize by l, write (b, N, c) f32 for the proj GEMM ----
#pragma unroll
  for (int r = 0; r < 8; ++r) {
    float inv = 1.0f / l_r[r];
    int row = row0 + mbase + r;
    float* op = o_ws + ((size_t)(b * 1024) + row) * 256 + h * 32;
    op[nloc]      = o0[r] * inv;
    op[nloc + 16] = o1[r] * inv;
  }
}

// ---------------------------------------------------------------------------
extern "C" void kernel_launch(void* const* d_in, const int* in_sizes, int n_in,
                              void* d_out, int out_size, void* d_ws, size_t ws_size,
                              hipStream_t stream) {
  const float* x    = (const float*)d_in[0];
  const float* q1_w = (const float*)d_in[1];
  const float* q1_b = (const float*)d_in[2];
  const float* q2_w = (const float*)d_in[3];
  const float* q2_b = (const float*)d_in[4];
  const float* k1_w = (const float*)d_in[5];
  const float* k1_b = (const float*)d_in[6];
  const float* k2_w = (const float*)d_in[7];
  const float* k2_b = (const float*)d_in[8];
  const float* v1_w = (const float*)d_in[9];
  const float* v1_b = (const float*)d_in[10];
  const float* v2_w = (const float*)d_in[11];
  const float* v2_b = (const float*)d_in[12];
  const float* p1_w = (const float*)d_in[13];
  const float* p1_b = (const float*)d_in[14];
  const float* p2_w = (const float*)d_in[15];
  const float* p2_b = (const float*)d_in[16];
  const float* lsc  = (const float*)d_in[17];
  const float* mw1  = (const float*)d_in[18];
  const float* mb1  = (const float*)d_in[19];
  const float* mw2  = (const float*)d_in[20];

  char* ws = (char*)d_ws;
  f16*   x3h   = (f16*)(ws);                          //  8 MB  (16384 x 256 f16)
  f16*   q_ws  = (f16*)(ws + ((size_t)8  << 20));     //  8 MB
  f16*   k_ws  = (f16*)(ws + ((size_t)16 << 20));     //  8 MB
  f16*   vt_ws = (f16*)(ws + ((size_t)24 << 20));     //  8 MB
  float* o_ws  = (float*)(ws + ((size_t)32 << 20));   // 16 MB  (16384 x 256 f32)
  float* tab   = (float*)(ws + ((size_t)48 << 20));   // 124 KB (8 x 3969 f32)

  transpose_kernel<<<16 * 256 * 1024 / 256, 256, 0, stream>>>(x, x3h);
  bias_table_kernel<<<(8 * 3969 + 255) / 256, 256, 0, stream>>>(mw1, mb1, mw2, tab);

  dim3 g(128);  // 16384 rows / 128
  half_gemm_kernel<0><<<g, 256, 0, stream>>>(x3h, nullptr, q1_w, q1_b, q_ws,  nullptr, 0);
  half_gemm_kernel<0><<<g, 256, 0, stream>>>(x3h, nullptr, q2_w, q2_b, q_ws,  nullptr, 1);
  half_gemm_kernel<0><<<g, 256, 0, stream>>>(x3h, nullptr, k1_w, k1_b, k_ws,  nullptr, 0);
  half_gemm_kernel<0><<<g, 256, 0, stream>>>(x3h, nullptr, k2_w, k2_b, k_ws,  nullptr, 1);
  half_gemm_kernel<1><<<g, 256, 0, stream>>>(x3h, nullptr, v1_w, v1_b, vt_ws, nullptr, 0);
  half_gemm_kernel<1><<<g, 256, 0, stream>>>(x3h, nullptr, v2_w, v2_b, vt_ws, nullptr, 1);

  attn_kernel<<<dim3(8, 8, 16), 256, 0, stream>>>(q_ws, k_ws, vt_ws, tab, lsc, o_ws);

  half_gemm_kernel<2><<<g, 256, 0, stream>>>(nullptr, o_ws, p1_w, p1_b, nullptr, (float*)d_out, 0);
  half_gemm_kernel<2><<<g, 256, 0, stream>>>(nullptr, o_ws, p2_w, p2_b, nullptr, (float*)d_out, 1);
}
